// GAT_6846177869902
// MI455X (gfx1250) — matrix operand order repaired and checked by reading, not compiled, not run on hardware
//
#include <hip/hip_runtime.h>
#include <hip/hip_bf16.h>

// ---------------- problem constants ----------------
#define N_NODES   50000
#define N_EDGES   1250000
#define E_TOT     (N_EDGES + N_NODES)   // edges + self loops
#define F_IN      512
#define H1C       64                    // heads1*ch1
#define HEADS1    8
#define CH1       8
#define NCLS      40
#define NCLS_PAD  48
#define MT1       (N_NODES / 16)        // 3125 M-tiles (exact)
#define KT1       (F_IN / 32)           // 16
#define NT1       (H1C / 16)            // 4
#define KT2       (H1C / 32)            // 2
#define NT2       (NCLS_PAD / 16)       // 3
#define NEG_SLOPE 0.2f
#define EPS_F     1e-16f

typedef __attribute__((ext_vector_type(16))) _Float16 v16h;
typedef __attribute__((ext_vector_type(8)))  float    v8f;
typedef __attribute__((ext_vector_type(4)))  unsigned int uint32x4;
typedef __attribute__((ext_vector_type(8)))  int          int32x8;
typedef __attribute__((ext_vector_type(4)))  int          int32x4;

// ---------------- WMMA per-lane layout helpers (CDNA5 ISA 7.12.2) ----------------
__device__ __forceinline__ int a_kk(int j, int half) {
  int i = j >> 1, lo = j & 1;
  return ((i < 4) ? (2 * i) : (2 * i + 8)) + 8 * half + lo;
}
__device__ __forceinline__ int b_kk(int j, int half) {
  int i = j >> 1, lo = j & 1;
  return ((i < 4) ? (2 * i) : (2 * i + 8)) + 8 * half + lo;
}

// LDS byte offset of a __shared__ object (low 32 bits of generic pointer = AS(3) offset)
__device__ __forceinline__ unsigned lds_off(const void* p) {
  return (unsigned)(unsigned long long)p;
}

// ---------------- TDM: one DMA of `n8` 8-byte elements global -> LDS ----------------
// D# per CDNA5 ISA ch.8: group0 {count, lds_addr, global_addr, type=2},
// group1 {data_size=8B, tensor_dim0=n8, tensor_dim1=1, tile_dim0=n8, tile_dim1=1, stride=n8}
__device__ __forceinline__ void tdm_load_to_lds(const void* gsrc, void* lds, unsigned n8) {
  unsigned long long ga = (unsigned long long)gsrc;
  unsigned la = lds_off(lds);
  uint32x4 g0;
  g0.x = 0x1u;                                           // count=1, user mode
  g0.y = la;                                             // lds_addr
  g0.z = (unsigned)(ga & 0xFFFFFFFFu);                   // global_addr[31:0]
  g0.w = (unsigned)((ga >> 32) & 0x1FFFFFFu) | (2u << 30); // global_addr[56:32] | type=2
  int32x8 g1;
  g1[0] = (int)(3u << 16);                               // data_size=3 (8 bytes)
  g1[1] = (int)((n8 & 0xFFFFu) << 16);                   // tensor_dim0[15:0] @ bits[63:48]
  g1[2] = (int)(((n8 >> 16) & 0xFFFFu) | (1u << 16));    // tensor_dim0[31:16] | tensor_dim1=1
  g1[3] = (int)((n8 & 0xFFFFu) << 16);                   // tile_dim0 @ bits[127:112]
  g1[4] = 1;                                             // tile_dim1=1, tile_dim2=0
  g1[5] = (int)n8;                                       // tensor_dim0_stride[31:0]
  g1[6] = 0;
  g1[7] = 0;
  int32x4 z4 = {0, 0, 0, 0};
#if __clang_major__ >= 23
  int32x8 z8 = {0, 0, 0, 0, 0, 0, 0, 0};
  __builtin_amdgcn_tensor_load_to_lds(g0, g1, z4, z4, z8, 0);
#else
  __builtin_amdgcn_tensor_load_to_lds(g0, g1, z4, z4, 0);
#endif
}

// ---------------- async global->LDS copy (per-lane b128), ASYNCcnt tracked ----------------
__device__ __forceinline__ void async_copy_b128(const void* gsrc, void* lds) {
  unsigned long long ga = (unsigned long long)gsrc;
  unsigned la = lds_off(lds);
  asm volatile("global_load_async_to_lds_b128 %0, %1, off"
               :: "v"(la), "v"(ga) : "memory");
}
__device__ __forceinline__ void wait_asynccnt0() {
  asm volatile("s_wait_asynccnt 0x0" ::: "memory");
}

// float atomic max via monotone integer punning (works with -inf init)
__device__ __forceinline__ void atomic_max_f32(float* a, float v) {
  if (v >= 0.0f) atomicMax((int*)a, __float_as_int(v));
  else           atomicMin((unsigned int*)a, __float_as_uint(v));
}

__device__ __forceinline__ void edge_nodes(const int* __restrict__ ei, int e, int& s, int& d) {
  if (e < N_EDGES) { s = ei[e]; d = ei[N_EDGES + e]; }
  else             { s = d = e - N_EDGES; }
}

// ---------------- utility ----------------
__global__ void fill_f32(float* __restrict__ p, float v, int n) {
  int i = blockIdx.x * blockDim.x + threadIdx.x;
  if (i < n) p[i] = v;
}

// ---------------- layer 1: swizzle x -> WMMA A layout (f16) ----------------
__global__ void swizzleA1(const float* __restrict__ x, _Float16* __restrict__ As) {
  int tid = blockIdx.x * blockDim.x + threadIdx.x;
  if (tid >= MT1 * KT1 * 32 * 16) return;
  int j = tid & 15, lane = (tid >> 4) & 31, kt = (tid >> 9) & 15, mt = tid >> 13;
  int half = lane >> 4, m = lane & 15;
  int row = mt * 16 + m;
  int col = kt * 32 + a_kk(j, half);
  As[tid] = (_Float16)x[(size_t)row * F_IN + col];
}

__global__ void swizzleB1(const float* __restrict__ W, _Float16* __restrict__ Bs) {
  int tid = blockIdx.x * blockDim.x + threadIdx.x;
  if (tid >= KT1 * NT1 * 32 * 16) return;
  int j = tid & 15, lane = (tid >> 4) & 31, nt = (tid >> 9) & 3, kt = tid >> 11;
  int half = lane >> 4, n = lane & 15;
  int r = kt * 32 + b_kk(j, half);
  int c = nt * 16 + n;
  Bs[tid] = (_Float16)W[(size_t)r * H1C + c];
}

// ---------------- layer 1 GEMM: h1 = x @ W1 ----------------
// B panel (64 KB) staged once per workgroup into LDS by a single TDM DMA;
// A panel streamed global->VGPR; inner loop: ds_load_b128 (B) + WMMA.
__global__ void gemm1(const _Float16* __restrict__ As, const _Float16* __restrict__ Bs,
                      float* __restrict__ h1) {
  __shared__ _Float16 Bsh[KT1 * NT1 * 32 * 16];      // 64 KB
  if (threadIdx.x < 32) {                            // wave 0 issues the DMA
    tdm_load_to_lds(Bs, Bsh, (KT1 * NT1 * 32 * 16 * 2) / 8);
    __builtin_amdgcn_s_wait_tensorcnt(0);
  }
  __syncthreads();

  int wave = (blockIdx.x * blockDim.x + threadIdx.x) >> 5;
  int lane = threadIdx.x & 31;
  if (wave < MT1) {
    int half = lane >> 4, nlo = lane & 15;

    v8f acc0 = {}, acc1 = {}, acc2 = {}, acc3 = {};
    const v16h* Ab = (const v16h*)As + (size_t)wave * (KT1 * 32) + lane;
    const v16h* Bl = (const v16h*)Bsh + lane;

#pragma unroll
    for (int kt = 0; kt < KT1; ++kt) {
      v16h a  = Ab[kt * 32];
      __builtin_prefetch(&Ab[(kt + 1) * 32], 0, 0);  // global_prefetch_b8
      v16h b0 = Bl[(kt * NT1 + 0) * 32];             // ds_load_b128 x2 each
      v16h b1 = Bl[(kt * NT1 + 1) * 32];
      v16h b2 = Bl[(kt * NT1 + 2) * 32];
      v16h b3 = Bl[(kt * NT1 + 3) * 32];
      acc0 = __builtin_amdgcn_wmma_f32_16x16x32_f16(false, a, false, b0, (short)0, acc0, false, false);
      acc1 = __builtin_amdgcn_wmma_f32_16x16x32_f16(false, a, false, b1, (short)0, acc1, false, false);
      acc2 = __builtin_amdgcn_wmma_f32_16x16x32_f16(false, a, false, b2, (short)0, acc2, false, false);
      acc3 = __builtin_amdgcn_wmma_f32_16x16x32_f16(false, a, false, b3, (short)0, acc3, false, false);
    }

    float* out = h1 + (size_t)wave * 16 * H1C;
#pragma unroll
    for (int r = 0; r < 8; ++r) {
      int row = r + 8 * half;                        // C/D layout: vgpr r, lane half
      float* o = out + (size_t)row * H1C + nlo;
      o[0]  = acc0[r];
      o[16] = acc1[r];
      o[32] = acc2[r];
      o[48] = acc3[r];
    }
  }
}

// ---------------- attention coefficients: alpha_s/alpha_d per (node, head) ----------------
__global__ void attn1(const float* __restrict__ h1, const float* __restrict__ att_s,
                      const float* __restrict__ att_d, float* __restrict__ as_,
                      float* __restrict__ ad_) {
  int tid = blockIdx.x * blockDim.x + threadIdx.x;
  if (tid >= N_NODES * HEADS1) return;
  int n = tid >> 3, hh = tid & 7;
  const float* hp = h1 + (size_t)n * H1C + hh * CH1;
  float s = 0.f, d = 0.f;
#pragma unroll
  for (int c = 0; c < CH1; ++c) {
    s += hp[c] * att_s[hh * CH1 + c];
    d += hp[c] * att_d[hh * CH1 + c];
  }
  as_[tid] = s;
  ad_[tid] = d;
}

// ---------------- layer 1 edge passes ----------------
__global__ void edge_max1(const int* __restrict__ ei, const float* __restrict__ as_,
                          const float* __restrict__ ad_, float* __restrict__ m1) {
  int e = blockIdx.x * blockDim.x + threadIdx.x;
  if (e >= E_TOT) return;
  int s, d; edge_nodes(ei, e, s, d);
#pragma unroll
  for (int hh = 0; hh < HEADS1; ++hh) {
    float v = as_[s * HEADS1 + hh] + ad_[d * HEADS1 + hh];
    v = v > 0.f ? v : NEG_SLOPE * v;
    atomic_max_f32(&m1[d * HEADS1 + hh], v);
  }
}

__global__ void edge_sum1(const int* __restrict__ ei, const float* __restrict__ as_,
                          const float* __restrict__ ad_, const float* __restrict__ m1,
                          float* __restrict__ s1) {
  int e = blockIdx.x * blockDim.x + threadIdx.x;
  if (e >= E_TOT) return;
  int s, d; edge_nodes(ei, e, s, d);
#pragma unroll
  for (int hh = 0; hh < HEADS1; ++hh) {
    float v = as_[s * HEADS1 + hh] + ad_[d * HEADS1 + hh];
    v = v > 0.f ? v : NEG_SLOPE * v;
    atomicAdd(&s1[d * HEADS1 + hh], __expf(v - m1[d * HEADS1 + hh]));
  }
}

__global__ void edge_msg1(const int* __restrict__ ei, const float* __restrict__ as_,
                          const float* __restrict__ ad_, const float* __restrict__ m1,
                          const float* __restrict__ s1, const float* __restrict__ h1,
                          float* __restrict__ out1) {
  int tid = blockIdx.x * blockDim.x + threadIdx.x;
  if (tid >= E_TOT * HEADS1) return;
  int e = tid >> 3, hh = tid & 7;
  int s, d; edge_nodes(ei, e, s, d);
  float v = as_[s * HEADS1 + hh] + ad_[d * HEADS1 + hh];
  v = v > 0.f ? v : NEG_SLOPE * v;
  float alpha = __expf(v - m1[d * HEADS1 + hh]) / (s1[d * HEADS1 + hh] + EPS_F);
  const float* hp = h1 + (size_t)s * H1C + hh * CH1;
  float* op = out1 + (size_t)d * H1C + hh * CH1;
#pragma unroll
  for (int c = 0; c < CH1; ++c) atomicAdd(&op[c], hp[c] * alpha);
}

// ---------------- bias + ELU fused with swizzle into layer-2 WMMA A layout ----------------
__global__ void elu_swizzleA2(const float* __restrict__ out1, const float* __restrict__ b1,
                              _Float16* __restrict__ As2) {
  int tid = blockIdx.x * blockDim.x + threadIdx.x;
  if (tid >= MT1 * KT2 * 32 * 16) return;
  int j = tid & 15, lane = (tid >> 4) & 31, kt = (tid >> 9) & 1, mt = tid >> 10;
  int half = lane >> 4, m = lane & 15;
  int row = mt * 16 + m;
  int col = kt * 32 + a_kk(j, half);
  float z = out1[(size_t)row * H1C + col] + b1[col];
  float e = z > 0.f ? z : (__expf(z) - 1.f);
  As2[tid] = (_Float16)e;
}

__global__ void swizzleB2(const float* __restrict__ W2, _Float16* __restrict__ Bs2) {
  int tid = blockIdx.x * blockDim.x + threadIdx.x;
  if (tid >= KT2 * NT2 * 32 * 16) return;
  int j = tid & 15, lane = (tid >> 4) & 31;
  int grp = tid >> 9;              // 0..5
  int nt = grp % NT2, kt = grp / NT2;
  int half = lane >> 4, n = lane & 15;
  int r = kt * 32 + b_kk(j, half);
  int c = nt * 16 + n;
  Bs2[tid] = (_Float16)((c < NCLS) ? W2[(size_t)r * NCLS + c] : 0.f);
}

// ---------------- layer 2 GEMM: h2 = elu(out1+b1) @ W2 ----------------
// B panel (6 KB) staged via async global->LDS b128 copies (ASYNCcnt path).
__global__ void gemm2(const _Float16* __restrict__ As2, const _Float16* __restrict__ Bs2,
                      float* __restrict__ h2) {
  __shared__ _Float16 Bsh[KT2 * NT2 * 32 * 16];     // 6 KB
  for (int i = threadIdx.x; i < (KT2 * NT2 * 32 * 16) / 8; i += blockDim.x) {
    async_copy_b128(Bs2 + i * 8, Bsh + i * 8);      // global_load_async_to_lds_b128
  }
  wait_asynccnt0();                                  // s_wait_asynccnt 0
  __syncthreads();

  int wave = (blockIdx.x * blockDim.x + threadIdx.x) >> 5;
  int lane = threadIdx.x & 31;
  if (wave < MT1) {
    int half = lane >> 4, nlo = lane & 15;

    v8f a0 = {}, a1 = {}, a2 = {};
    const v16h* Ab = (const v16h*)As2 + (size_t)wave * (KT2 * 32) + lane;
    const v16h* Bl = (const v16h*)Bsh + lane;

#pragma unroll
    for (int kt = 0; kt < KT2; ++kt) {
      v16h a  = Ab[kt * 32];
      v16h b0 = Bl[(kt * NT2 + 0) * 32];
      v16h b1 = Bl[(kt * NT2 + 1) * 32];
      v16h b2 = Bl[(kt * NT2 + 2) * 32];
      a0 = __builtin_amdgcn_wmma_f32_16x16x32_f16(false, a, false, b0, (short)0, a0, false, false);
      a1 = __builtin_amdgcn_wmma_f32_16x16x32_f16(false, a, false, b1, (short)0, a1, false, false);
      a2 = __builtin_amdgcn_wmma_f32_16x16x32_f16(false, a, false, b2, (short)0, a2, false, false);
    }

#pragma unroll
    for (int r = 0; r < 8; ++r) {
      int row = wave * 16 + r + 8 * half;
      float* o = h2 + (size_t)row * NCLS;
      o[nlo]      = a0[r];
      o[16 + nlo] = a1[r];
      if (nlo < NCLS - 32) o[32 + nlo] = a2[r];     // drop padded cols 40..47
    }
  }
}

// ---------------- layer 2 attention coefficients (1 head, 40 ch) ----------------
__global__ void attn2(const float* __restrict__ h2, const float* __restrict__ att_s,
                      const float* __restrict__ att_d, float* __restrict__ as_,
                      float* __restrict__ ad_) {
  int n = blockIdx.x * blockDim.x + threadIdx.x;
  if (n >= N_NODES) return;
  const float* hp = h2 + (size_t)n * NCLS;
  float s = 0.f, d = 0.f;
#pragma unroll
  for (int c = 0; c < NCLS; ++c) { s += hp[c] * att_s[c]; d += hp[c] * att_d[c]; }
  as_[n] = s;
  ad_[n] = d;
}

// ---------------- layer 2 edge passes (single head) ----------------
__global__ void edge_max2(const int* __restrict__ ei, const float* __restrict__ as_,
                          const float* __restrict__ ad_, float* __restrict__ m2) {
  int e = blockIdx.x * blockDim.x + threadIdx.x;
  if (e >= E_TOT) return;
  int s, d; edge_nodes(ei, e, s, d);
  float v = as_[s] + ad_[d];
  v = v > 0.f ? v : NEG_SLOPE * v;
  atomic_max_f32(&m2[d], v);
}

__global__ void edge_sum2(const int* __restrict__ ei, const float* __restrict__ as_,
                          const float* __restrict__ ad_, const float* __restrict__ m2,
                          float* __restrict__ s2) {
  int e = blockIdx.x * blockDim.x + threadIdx.x;
  if (e >= E_TOT) return;
  int s, d; edge_nodes(ei, e, s, d);
  float v = as_[s] + ad_[d];
  v = v > 0.f ? v : NEG_SLOPE * v;
  atomicAdd(&s2[d], __expf(v - m2[d]));
}

__global__ void edge_msg2(const int* __restrict__ ei, const float* __restrict__ as_,
                          const float* __restrict__ ad_, const float* __restrict__ m2,
                          const float* __restrict__ s2, const float* __restrict__ h2,
                          float* __restrict__ out2) {
  int e = blockIdx.x * blockDim.x + threadIdx.x;
  if (e >= E_TOT) return;
  int s, d; edge_nodes(ei, e, s, d);
  float v = as_[s] + ad_[d];
  v = v > 0.f ? v : NEG_SLOPE * v;
  float alpha = __expf(v - m2[d]) / (s2[d] + EPS_F);
  const float* hp = h2 + (size_t)s * NCLS;
  float* op = out2 + (size_t)d * NCLS;
#pragma unroll
  for (int c = 0; c < NCLS; ++c) atomicAdd(&op[c], hp[c] * alpha);
}

// ---------------- final bias + log_softmax ----------------
__global__ void logsoftmax(const float* __restrict__ out2, const float* __restrict__ b2,
                           float* __restrict__ y) {
  int n = blockIdx.x * blockDim.x + threadIdx.x;
  if (n >= N_NODES) return;
  float z[NCLS];
  float m = -__builtin_inff();
#pragma unroll
  for (int c = 0; c < NCLS; ++c) {
    z[c] = out2[(size_t)n * NCLS + c] + b2[c];
    m = fmaxf(m, z[c]);
  }
  float sum = 0.f;
#pragma unroll
  for (int c = 0; c < NCLS; ++c) sum += __expf(z[c] - m);
  float l = __logf(sum);
#pragma unroll
  for (int c = 0; c < NCLS; ++c) y[(size_t)n * NCLS + c] = z[c] - m - l;
}

// ---------------- host-side launcher ----------------
extern "C" void kernel_launch(void* const* d_in, const int* in_sizes, int n_in,
                              void* d_out, int out_size, void* d_ws, size_t ws_size,
                              hipStream_t stream) {
  const float* x      = (const float*)d_in[0];
  const int*   ei     = (const int*)d_in[1];   // edge_index [2, E]
  const float* W1     = (const float*)d_in[2];
  const float* att_s1 = (const float*)d_in[3];
  const float* att_d1 = (const float*)d_in[4];
  const float* b1     = (const float*)d_in[5];
  const float* W2     = (const float*)d_in[6];
  const float* att_s2 = (const float*)d_in[7];
  const float* att_d2 = (const float*)d_in[8];
  const float* b2     = (const float*)d_in[9];
  float* y = (float*)d_out;

  size_t off = 0;
  auto take = [&](size_t bytes) -> void* {
    void* p = (char*)d_ws + off;
    off = (off + bytes + 255) & ~(size_t)255;
    return p;
  };

  _Float16* As1 = (_Float16*)take((size_t)MT1 * KT1 * 512 * sizeof(_Float16)); // 51.2 MB
  _Float16* Bs1 = (_Float16*)take((size_t)KT1 * NT1 * 512 * sizeof(_Float16));
  float* h1   = (float*)take((size_t)N_NODES * H1C * sizeof(float));
  float* as1  = (float*)take((size_t)N_NODES * HEADS1 * sizeof(float));
  float* ad1  = (float*)take((size_t)N_NODES * HEADS1 * sizeof(float));
  float* m1   = (float*)take((size_t)N_NODES * HEADS1 * sizeof(float));
  float* s1   = (float*)take((size_t)N_NODES * HEADS1 * sizeof(float));
  float* out1 = (float*)take((size_t)N_NODES * H1C * sizeof(float));
  _Float16* As2 = (_Float16*)take((size_t)MT1 * KT2 * 512 * sizeof(_Float16));
  _Float16* Bs2 = (_Float16*)take((size_t)KT2 * NT2 * 512 * sizeof(_Float16));
  float* h2   = (float*)take((size_t)N_NODES * NCLS * sizeof(float));
  float* as2  = (float*)take((size_t)N_NODES * sizeof(float));
  float* ad2  = (float*)take((size_t)N_NODES * sizeof(float));
  float* m2   = (float*)take((size_t)N_NODES * sizeof(float));
  float* s2   = (float*)take((size_t)N_NODES * sizeof(float));
  float* out2 = (float*)take((size_t)N_NODES * NCLS * sizeof(float));

  auto nb = [](long long n, int b) { return (unsigned)((n + b - 1) / b); };
  const float NEG_INF = -__builtin_inff();

  // ---- layer 1 ----
  swizzleA1<<<nb((long long)MT1 * KT1 * 512, 256), 256, 0, stream>>>(x, As1);
  swizzleB1<<<nb((long long)KT1 * NT1 * 512, 256), 256, 0, stream>>>(W1, Bs1);
  gemm1<<<nb((long long)MT1 * 32, 128), 128, 0, stream>>>(As1, Bs1, h1);
  attn1<<<nb((long long)N_NODES * HEADS1, 256), 256, 0, stream>>>(h1, att_s1, att_d1, as1, ad1);

  fill_f32<<<nb(N_NODES * HEADS1, 256), 256, 0, stream>>>(m1, NEG_INF, N_NODES * HEADS1);
  fill_f32<<<nb(N_NODES * HEADS1, 256), 256, 0, stream>>>(s1, 0.f, N_NODES * HEADS1);
  fill_f32<<<nb((long long)N_NODES * H1C, 256), 256, 0, stream>>>(out1, 0.f, N_NODES * H1C);

  edge_max1<<<nb(E_TOT, 256), 256, 0, stream>>>(ei, as1, ad1, m1);
  edge_sum1<<<nb(E_TOT, 256), 256, 0, stream>>>(ei, as1, ad1, m1, s1);
  edge_msg1<<<nb((long long)E_TOT * HEADS1, 256), 256, 0, stream>>>(ei, as1, ad1, m1, s1, h1, out1);

  // ---- layer 2 ----
  elu_swizzleA2<<<nb((long long)MT1 * KT2 * 512, 256), 256, 0, stream>>>(out1, b1, As2);
  swizzleB2<<<nb((long long)KT2 * NT2 * 512, 256), 256, 0, stream>>>(W2, Bs2);
  gemm2<<<nb((long long)MT1 * 32, 128), 128, 0, stream>>>(As2, Bs2, h2);
  attn2<<<nb(N_NODES, 256), 256, 0, stream>>>(h2, att_s2, att_d2, as2, ad2);

  fill_f32<<<nb(N_NODES, 256), 256, 0, stream>>>(m2, NEG_INF, N_NODES);
  fill_f32<<<nb(N_NODES, 256), 256, 0, stream>>>(s2, 0.f, N_NODES);
  fill_f32<<<nb((long long)N_NODES * NCLS, 256), 256, 0, stream>>>(out2, 0.f, N_NODES * NCLS);

  edge_max2<<<nb(E_TOT, 256), 256, 0, stream>>>(ei, as2, ad2, m2);
  edge_sum2<<<nb(E_TOT, 256), 256, 0, stream>>>(ei, as2, ad2, m2, s2);
  edge_msg2<<<nb(E_TOT, 256), 256, 0, stream>>>(ei, as2, ad2, m2, s2, h2, out2);

  logsoftmax<<<nb(N_NODES, 256), 256, 0, stream>>>(out2, b2, y);
}
